// CausalMHA_48180943127225
// MI455X (gfx1250) — matrix operand order repaired and checked by reading, not compiled, not run on hardware
//
#include <hip/hip_runtime.h>
#include <stdint.h>

#define DM    1024
#define HEADS 16
#define DK    64
#define BATCH 2
#define SEQ   2048
#define MROWS (BATCH*SEQ)   // 4096

typedef __attribute__((ext_vector_type(16))) __bf16 bf16x16;
typedef __attribute__((ext_vector_type(8)))  __bf16 bf16x8;
typedef __attribute__((ext_vector_type(8)))  float  f32x8;

union BF16Frag { bf16x16 v; bf16x8 h[2]; };

__device__ __forceinline__ unsigned short f2bf(float f) {
  unsigned int u = __float_as_uint(f);
  u += 0x7fffu + ((u >> 16) & 1u);           // round-to-nearest-even
  return (unsigned short)(u >> 16);
}
__device__ __forceinline__ float bf2f(unsigned short b) {
  return __uint_as_float(((unsigned int)b) << 16);
}

// ---------------- fp32 -> bf16 convert ----------------
__global__ void k_f32_to_bf16(const float* __restrict__ in,
                              unsigned short* __restrict__ out, int n) {
  int i = blockIdx.x * blockDim.x + threadIdx.x;
  if (i < n) out[i] = f2bf(in[i]);
}

// ---------------- WMMA GEMM: Y[m][n] = sum_k A[m][k] * B[n][k] ----------------
// M=4096, N=K=1024. One wave computes a 32x64 tile: two A-row blocks share each
// B fragment -> 8 WMMA per 12 b128 loads per k=32 step (~21 FLOP/byte).
__global__ void __launch_bounds__(128)
k_gemm_bf16(const unsigned short* __restrict__ A,
            const unsigned short* __restrict__ Bw,
            unsigned short* __restrict__ outb,
            float* __restrict__ outf) {
  const int wid  = blockIdx.x * 4 + (threadIdx.x >> 5);
  const int lane = threadIdx.x & 31;
  const int half = lane >> 4, ln = lane & 15;
  const int rowTiles = MROWS / 32;          // 128
  const int rt = wid % rowTiles;
  const int cg = wid / rowTiles;            // 0..15
  const int m0 = rt * 32, n0 = cg * 64;

  f32x8 c0[4] = {};
  f32x8 c1[4] = {};
  const unsigned short* arow0 = A + (size_t)(m0 + ln) * DM;
  const unsigned short* arow1 = A + (size_t)(m0 + 16 + ln) * DM;
  for (int k = 0; k < DM; k += 32) {
    BF16Frag a0, a1;
    a0.h[0] = *(const bf16x8*)(arow0 + k + 16 * half);       // K = k+16h .. +7
    a0.h[1] = *(const bf16x8*)(arow0 + k + 16 * half + 8);   // K = k+16h+8 .. +7
    a1.h[0] = *(const bf16x8*)(arow1 + k + 16 * half);
    a1.h[1] = *(const bf16x8*)(arow1 + k + 16 * half + 8);
#pragma unroll
    for (int t = 0; t < 4; ++t) {
      const unsigned short* brow =
          Bw + (size_t)(n0 + 16 * t + ln) * DM + k + 16 * half;
      bf16x16 b = *(const bf16x16*)brow;                     // K contiguous 16
      c0[t] = __builtin_amdgcn_wmma_f32_16x16x32_bf16(
          false, a0.v, false, b, (short)0, c0[t], false, false);
      c1[t] = __builtin_amdgcn_wmma_f32_16x16x32_bf16(
          false, a1.v, false, b, (short)0, c1[t], false, false);
    }
  }
#pragma unroll
  for (int t = 0; t < 4; ++t)
#pragma unroll
    for (int r = 0; r < 8; ++r) {
      int col  = n0 + 16 * t + ln;
      int row0 = m0 + r + 8 * half;
      int row1 = row0 + 16;
      float v0 = c0[t][r];
      float v1 = c1[t][r];
      if (outf) {
        outf[(size_t)row0 * DM + col] = v0;
        outf[(size_t)row1 * DM + col] = v1;
      } else {
        outb[(size_t)row0 * DM + col] = f2bf(v0);
        outb[(size_t)row1 * DM + col] = f2bf(v1);
      }
    }
}

// ---------------- RoPE in-place on bf16 [B,S,H,DK], one thread per pair ----------------
__global__ void k_rope_inplace(unsigned short* __restrict__ qk,
                               const int* __restrict__ pos, int npairs) {
  int i = blockIdx.x * blockDim.x + threadIdx.x;
  if (i >= npairs) return;
  int j  = i & 31;            // freq index (DK/2 = 32)
  int t1 = i >> 5;
  int h  = t1 & 15;
  int t2 = t1 >> 4;
  int s  = t2 & 2047;
  int b  = t2 >> 11;
  float p  = (float)pos[s];
  float fr = __expf(-(float)j * (9.210340371976184f / 32.0f)); // 10000^(-j/32)
  float an = p * fr, sn, cs;
  __sincosf(an, &sn, &cs);
  size_t base = (size_t)(b * SEQ + s) * DM + h * DK + 2 * j;
  float x1 = bf2f(qk[base]), x2 = bf2f(qk[base + 1]);
  qk[base]     = f2bf(x1 * cs - x2 * sn);
  qk[base + 1] = f2bf(x1 * sn + x2 * cs);
}

// ---------------- V transpose: [B,S,H,DK] -> [B,H,DK,S] ----------------
__global__ void k_vtrans(const unsigned short* __restrict__ V,
                         unsigned short* __restrict__ Vt, int n) {
  int i = blockIdx.x * blockDim.x + threadIdx.x;
  if (i >= n) return;
  int d  = i & 63;
  int t  = i >> 6;
  int h  = t & 15;
  int t2 = t >> 4;
  int s  = t2 & 2047;
  int b  = t2 >> 11;
  Vt[((size_t)((b * HEADS + h) * DK + d)) * SEQ + s] = V[i];
}

// ---------------- Flash attention: one wave per (b,h,16-query tile) ----------------
__global__ void __launch_bounds__(128)
k_flash(const unsigned short* __restrict__ Q,
        const unsigned short* __restrict__ K,
        const unsigned short* __restrict__ Vt,
        unsigned short* __restrict__ O) {
  __shared__ unsigned short Pl[4][16 * 32];   // per-wave P staging (16x32 bf16)
  const int warp = threadIdx.x >> 5;
  const int lane = threadIdx.x & 31;
  const int half = lane >> 4, ln = lane & 15;
  const int wid  = blockIdx.x * 4 + warp;
  const int qt   = wid & 127;                 // S/16 = 128 tiles
  const int bh   = wid >> 7;
  const int b    = bh >> 4, h = bh & 15;
  const int q0   = qt * 16;

  // Q fragments (16 x 64 head-dim -> two K=32 chunks)
  const unsigned short* qrow = Q + (size_t)(b * SEQ + q0 + ln) * DM + h * DK;
  BF16Frag aq0, aq1;
  aq0.h[0] = *(const bf16x8*)(qrow + 16 * half);
  aq0.h[1] = *(const bf16x8*)(qrow + 8 + 16 * half);
  aq1.h[0] = *(const bf16x8*)(qrow + 32 + 16 * half);
  aq1.h[1] = *(const bf16x8*)(qrow + 40 + 16 * half);

  float mrow[8], lrow[8];
  f32x8 o[4] = {};
#pragma unroll
  for (int r = 0; r < 8; ++r) { mrow[r] = -__builtin_inff(); lrow[r] = 0.f; }

  unsigned short* pl = Pl[warp];
  const int kend = q0 + 16;                   // causal: keys 0..q0+15
  for (int kk = 0; kk < kend; kk += 32) {
    float sv[2][8];
#pragma unroll
    for (int g = 0; g < 2; ++g) {
      int kc = kk + 16 * g;
      const unsigned short* krow = K + (size_t)(b * SEQ + kc + ln) * DM + h * DK;
      bf16x16 b0 = *(const bf16x16*)(krow + 16 * half);
      bf16x16 b1 = *(const bf16x16*)(krow + 32 + 16 * half);
      f32x8 s = {};
      s = __builtin_amdgcn_wmma_f32_16x16x32_bf16(false, aq0.v, false, b0,
                                                  (short)0, s, false, false);
      s = __builtin_amdgcn_wmma_f32_16x16x32_bf16(false, aq1.v, false, b1,
                                                  (short)0, s, false, false);
#pragma unroll
      for (int r = 0; r < 8; ++r) {
        int qpos = q0 + r + 8 * half;
        int kpos = kc + ln;
        sv[g][r] = (kpos <= qpos) ? s[r] * 0.125f : -__builtin_inff();
      }
    }
    // online softmax per row (row = r + 8*half, spread over 16 lanes of a half)
#pragma unroll
    for (int r = 0; r < 8; ++r) {
      float mx = fmaxf(sv[0][r], sv[1][r]);
      mx = fmaxf(mx, __shfl_xor(mx, 1));
      mx = fmaxf(mx, __shfl_xor(mx, 2));
      mx = fmaxf(mx, __shfl_xor(mx, 4));
      mx = fmaxf(mx, __shfl_xor(mx, 8));
      float mn    = fmaxf(mrow[r], mx);
      float alpha = __expf(mrow[r] - mn);
      float p0    = __expf(sv[0][r] - mn);
      float p1    = __expf(sv[1][r] - mn);
      float rs = p0 + p1;
      rs += __shfl_xor(rs, 1);
      rs += __shfl_xor(rs, 2);
      rs += __shfl_xor(rs, 4);
      rs += __shfl_xor(rs, 8);
      lrow[r] = lrow[r] * alpha + rs;
      mrow[r] = mn;
#pragma unroll
      for (int t = 0; t < 4; ++t) o[t][r] *= alpha;
      pl[(r + 8 * half) * 32 + ln]      = f2bf(p0);
      pl[(r + 8 * half) * 32 + 16 + ln] = f2bf(p1);
    }
    // reload P in A-layout (same-wave LDS ordering via DScnt)
    BF16Frag pa;
    pa.h[0] = *(const bf16x8*)(pl + ln * 32 + 16 * half);
    pa.h[1] = *(const bf16x8*)(pl + ln * 32 + 8 + 16 * half);
#pragma unroll
    for (int t = 0; t < 4; ++t) {
      const unsigned short* vrow =
          Vt + (size_t)((b * HEADS + h) * DK + 16 * t + ln) * SEQ + kk + 16 * half;
      bf16x16 vb = *(const bf16x16*)vrow;
      o[t] = __builtin_amdgcn_wmma_f32_16x16x32_bf16(
          false, pa.v, false, vb, (short)0, o[t], false, false);
    }
  }
#pragma unroll
  for (int t = 0; t < 4; ++t)
#pragma unroll
    for (int r = 0; r < 8; ++r) {
      int qpos = q0 + r + 8 * half;
      O[(size_t)(b * SEQ + qpos) * DM + h * DK + 16 * t + ln] =
          f2bf(o[t][r] / lrow[r]);
    }
}

// ---------------- host launch ----------------
extern "C" void kernel_launch(void* const* d_in, const int* in_sizes, int n_in,
                              void* d_out, int out_size, void* d_ws, size_t ws_size,
                              hipStream_t stream) {
  const float* x  = (const float*)d_in[0];
  const float* Wq = (const float*)d_in[1];
  const float* Wk = (const float*)d_in[2];
  const float* Wv = (const float*)d_in[3];
  const float* Wo = (const float*)d_in[4];
  const int*   tok = (const int*)d_in[5];
  float* out = (float*)d_out;

  char* ws = (char*)d_ws;
  unsigned short* xb  = (unsigned short*)(ws);                       // 8 MB
  unsigned short* wqb = (unsigned short*)(ws + ((size_t)8  << 20));  // 2 MB
  unsigned short* wkb = (unsigned short*)(ws + ((size_t)10 << 20));
  unsigned short* wvb = (unsigned short*)(ws + ((size_t)12 << 20));
  unsigned short* wob = (unsigned short*)(ws + ((size_t)14 << 20));
  unsigned short* Qb  = (unsigned short*)(ws + ((size_t)16 << 20));  // 8 MB
  unsigned short* Kb  = (unsigned short*)(ws + ((size_t)24 << 20));
  unsigned short* Vb  = (unsigned short*)(ws + ((size_t)32 << 20));
  unsigned short* Vtb = (unsigned short*)(ws + ((size_t)40 << 20));
  unsigned short* Ab  = (unsigned short*)(ws + ((size_t)48 << 20));

  const int NX = BATCH * SEQ * DM;   // 4M
  const int NW = DM * DM;            // 1M
  k_f32_to_bf16<<<(NX + 255) / 256, 256, 0, stream>>>(x,  xb,  NX);
  k_f32_to_bf16<<<(NW + 255) / 256, 256, 0, stream>>>(Wq, wqb, NW);
  k_f32_to_bf16<<<(NW + 255) / 256, 256, 0, stream>>>(Wk, wkb, NW);
  k_f32_to_bf16<<<(NW + 255) / 256, 256, 0, stream>>>(Wv, wvb, NW);
  k_f32_to_bf16<<<(NW + 255) / 256, 256, 0, stream>>>(Wo, wob, NW);

  const int gblocks = (MROWS / 32) * (DM / 64) / 4;   // 512 blocks x 4 waves
  k_gemm_bf16<<<gblocks, 128, 0, stream>>>(xb, wqb, Qb, nullptr);
  k_gemm_bf16<<<gblocks, 128, 0, stream>>>(xb, wkb, Kb, nullptr);
  k_gemm_bf16<<<gblocks, 128, 0, stream>>>(xb, wvb, Vb, nullptr);

  const int npairs = NX / 2;
  k_rope_inplace<<<(npairs + 255) / 256, 256, 0, stream>>>(Qb, tok, npairs);
  k_rope_inplace<<<(npairs + 255) / 256, 256, 0, stream>>>(Kb, tok, npairs);
  k_vtrans<<<(NX + 255) / 256, 256, 0, stream>>>(Vb, Vtb, NX);

  k_flash<<<(BATCH * HEADS * (SEQ / 16)) / 4, 128, 0, stream>>>(Qb, Kb, Vtb, Ab);

  k_gemm_bf16<<<gblocks, 128, 0, stream>>>(Ab, wob, nullptr, out);
}
